// Model_72275709657299
// MI455X (gfx1250) — compile-verified, matrix operand order checked by reading, not run on hardware
//
#include <hip/hip_runtime.h>
#include <math.h>

typedef __attribute__((ext_vector_type(16))) _Float16 v16h;
typedef __attribute__((ext_vector_type(8)))  _Float16 v8h;
typedef __attribute__((ext_vector_type(8)))  float    v8f;

#define B_TOT  4096
#define SEQ    96
#define PRED   24
#define WSTEP  120   // SEQ + PRED
#define HID    40
#define NS     20
#define INP    20
#define TILE_M 32    // two 16-row WMMA M-tiles per wave
#define NTHR   320   // 10 wave32s

#define LOG2E 1.4426950408889634f
#define LN2   0.6931471805599453f

// Hardware TRANS-unit math: v_exp_f32 (exp2), v_rcp_f32, v_log_f32 (log2).
__device__ __forceinline__ float fexp2(float x) { return __builtin_amdgcn_exp2f(x); }
__device__ __forceinline__ float frcp (float x) { return __builtin_amdgcn_rcpf(x); }
__device__ __forceinline__ float flog2(float x) { return __builtin_amdgcn_logf(x); }

__device__ __forceinline__ float sigm(float x) {
  return frcp(1.0f + fexp2(-LOG2E * x));
}
__device__ __forceinline__ float ftanh(float x) {
  // tanh(x) = 1 - 2/(exp(2x)+1)
  return __builtin_fmaf(-2.0f, frcp(1.0f + fexp2(2.0f * LOG2E * x)), 1.0f);
}
__device__ __forceinline__ float softplusf(float x) {
  return (x > 20.f) ? x : LN2 * flog2(1.0f + fexp2(LOG2E * x));
}

// LSTM cell pointwise update for one (row r, unit j); indices are loop-invariant.
__device__ __forceinline__ void cell_update(const float* zbuf, float* c, float* hf,
                                            _Float16* hh, int r, int j) {
  const float* z = zbuf + r * 160;
  float ig = sigm(z[j]);
  float fg = sigm(z[HID + j]);
  float gg = ftanh(z[2 * HID + j]);
  float og = sigm(z[3 * HID + j]);
  int u = r * HID + j;
  float cn = fg * c[u] + ig * gg;
  float hn = og * ftanh(cn);
  c[u] = cn; hf[u] = hn; hh[r * 64 + j] = (_Float16)hn;
}

// Build an A-fragment (16xK tile, K-tile of 32) from a padded f16 LDS tile.
// ISA 16-bit A layout: lane m=lane&15; halves 0..7 -> K=kb..kb+7,
// halves 8..15 -> K=kb+16..kb+23, kb = (lane<16?0:8) + 32*ktile.
__device__ __forceinline__ v16h ldsA(const _Float16* base, int strideH, int lane, int ktile) {
  int m  = lane & 15;
  int kb = ktile * 32 + ((lane & 16) ? 8 : 0);
  const _Float16* p = base + m * strideH + kb;
  v8h lo = *(const v8h*)(p);        // ds_load_b128
  v8h hi = *(const v8h*)(p + 16);   // ds_load_b128
  v16h a;
#pragma unroll
  for (int i = 0; i < 8; i++) { a[i] = lo[i]; a[i + 8] = hi[i]; }
  return a;
}

// Build a B-fragment (KxN slice, N = gate cols [col0,col0+16)) from global f32
// weights stored row-major [160][kdim]; z = x @ W^T so B[k][n] = W[col0+n][k].
__device__ __forceinline__ v16h mkB(const float* w, int kdim, int col0, int ktile, int lane) {
  int n  = col0 + (lane & 15);
  int kb = ktile * 32 + ((lane & 16) ? 8 : 0);
  v16h b;
#pragma unroll
  for (int i = 0; i < 16; i++) {
    int k = kb + ((i < 8) ? i : (i + 8));
    float v = (k < kdim) ? w[n * kdim + k] : 0.0f;
    b[i] = (_Float16)v;
  }
  return b;
}

#define WMMA_F16(A, B, C) \
  __builtin_amdgcn_wmma_f32_16x16x32_f16(false, (A), false, (B), (short)0, (C), false, false)

__global__ __launch_bounds__(NTHR) void lstm_deepar_kernel(
    const float* __restrict__ x_enc, const float* __restrict__ y_enc,
    const float* __restrict__ w_ih0, const float* __restrict__ w_hh0,
    const float* __restrict__ b_ih0, const float* __restrict__ b_hh0,
    const float* __restrict__ w_ih1, const float* __restrict__ w_hh1,
    const float* __restrict__ b_ih1, const float* __restrict__ b_hh1,
    const float* __restrict__ w_b0,  const float* __restrict__ b_b0,
    const float* __restrict__ w_g,   const float* __restrict__ b_g,
    float* __restrict__ out)
{
  __shared__ __align__(16) _Float16 xh [TILE_M * 32];   // x tile, K padded 20->32
  __shared__ __align__(16) _Float16 h1h[TILE_M * 64];   // h1 f16, K padded 40->64
  __shared__ __align__(16) _Float16 h2h[TILE_M * 64];   // h2 f16
  __shared__ __align__(16) float h1f[TILE_M * HID];
  __shared__ __align__(16) float h2f[TILE_M * HID];
  __shared__ __align__(16) float c1 [TILE_M * HID];
  __shared__ __align__(16) float c2 [TILE_M * HID];
  __shared__ __align__(16) float zbuf[TILE_M * 160];    // pre-activation gates
  __shared__ __align__(16) float whead[21 * 80];        // row0=w_b0, rows1..20=w_g
  __shared__ float bhead[21];

  const int tid  = threadIdx.x;
  const int lane = tid & 31;
  const int wave = tid >> 5;            // 0..9
  const int col0 = wave * 16;           // this wave's gate-column slice
  const int b0   = blockIdx.x * TILE_M; // batch tile base

  // ---- init LDS state (zeros also cover K-padding, kept zero forever) ----
  for (int i = tid; i < TILE_M * 32; i += NTHR) xh[i] = (_Float16)0.f;
  for (int i = tid; i < TILE_M * 64; i += NTHR) { h1h[i] = (_Float16)0.f; h2h[i] = (_Float16)0.f; }
  for (int i = tid; i < TILE_M * HID; i += NTHR) { h1f[i] = 0.f; h2f[i] = 0.f; c1[i] = 0.f; c2[i] = 0.f; }
  for (int i = tid; i < 21 * 80; i += NTHR) whead[i] = (i < 80) ? w_b0[i] : w_g[i - 80];
  if (tid < 21) bhead[tid] = (tid == 0) ? b_b0[0] : b_g[tid - 1];

  // ---- per-wave weight B-fragments, resident in VGPRs for whole kernel ----
  v16h Bx0   = mkB(w_ih0, INP, col0, 0, lane);
  v16h Bh0_0 = mkB(w_hh0, HID, col0, 0, lane);
  v16h Bh0_1 = mkB(w_hh0, HID, col0, 1, lane);
  v16h Bi1_0 = mkB(w_ih1, HID, col0, 0, lane);
  v16h Bi1_1 = mkB(w_ih1, HID, col0, 1, lane);
  v16h Bh1_0 = mkB(w_hh1, HID, col0, 0, lane);
  v16h Bh1_1 = mkB(w_hh1, HID, col0, 1, lane);

  const int   gcol  = col0 + (lane & 15);
  const float bias0 = b_ih0[gcol] + b_hh0[gcol];
  const float bias1 = b_ih1[gcol] + b_hh1[gcol];
  const int   mbase = (lane & 16) ? 8 : 0;   // C/D layout: VGPR r -> M = r + mbase
  const int   ncol  = lane & 15;

  // ---- loop-invariant per-thread index precomputation ----
  // x streaming: 32 rows x 20 feats = 640 = 2 elements/thread
  const int xr0 = tid / 20,          xk0 = tid % 20;
  const int xr1 = (tid + NTHR) / 20, xk1 = (tid + NTHR) % 20;
  const float* xsrc0 = x_enc + (size_t)(b0 + xr0) * SEQ  * 21 + xk0;
  const float* ysrc0 = y_enc + (size_t)(b0 + xr0) * PRED * 21 + xk0;
  const float* xsrc1 = x_enc + (size_t)(b0 + xr1) * SEQ  * 21 + xk1;
  const float* ysrc1 = y_enc + (size_t)(b0 + xr1) * PRED * 21 + xk1;
  _Float16* xdst0 = xh + xr0 * 32 + xk0;
  _Float16* xdst1 = xh + xr1 * 32 + xk1;
  // labels: threads 0..31
  const float* lxsrc = x_enc + (size_t)(b0 + tid) * SEQ  * 21 + 20;
  const float* lysrc = y_enc + (size_t)(b0 + tid) * PRED * 21 + 20;
  float* lout = out + (size_t)21 * WSTEP * B_TOT + (b0 + tid);
  // elementwise: exactly 4 units per thread (1280 = 4*320)
  int er[4], ej[4];
#pragma unroll
  for (int k = 0; k < 4; k++) { er[k] = (tid + k * NTHR) / HID; ej[k] = (tid + k * NTHR) % HID; }
  // heads: 672 = 2/thread + 32 extra (tid<32)
  int hr[3], hs[3];
#pragma unroll
  for (int k = 0; k < 3; k++) { hr[k] = (tid + k * NTHR) / 21; hs[k] = (tid + k * NTHR) % 21; }

  __syncthreads();

  for (int t = 0; t < WSTEP; ++t) {
    // ---- stream x tile + labels (uniform scalar select of source) ----
    {
      const float* s0 = (t < SEQ) ? (xsrc0 + t * 21) : (ysrc0 + (t - SEQ) * 21);
      const float* s1 = (t < SEQ) ? (xsrc1 + t * 21) : (ysrc1 + (t - SEQ) * 21);
      *xdst0 = (_Float16)(*s0);
      *xdst1 = (_Float16)(*s1);
      if (tid < TILE_M) {
        const float* sl = (t < SEQ) ? (lxsrc + t * 21) : (lysrc + (t - SEQ) * 21);
        lout[(size_t)t * B_TOT] = *sl;   // labels_t (not rolled)
      }
    }
    __syncthreads();

    // ---- layer0 GEMM: z0 = x @ Wih0^T + h1 @ Whh0^T + bias ----
    {
      v8f accA, accB;
#pragma unroll
      for (int r = 0; r < 8; r++) { accA[r] = bias0; accB[r] = bias0; }
      // M-tile 0 (rows 0..15) and M-tile 1 (rows 16..31): two independent chains
      v16h axA = ldsA(xh,       32, lane, 0);
      v16h axB = ldsA(xh  + 16*32, 32, lane, 0);
      v16h a0A = ldsA(h1h,      64, lane, 0);
      v16h a0B = ldsA(h1h + 16*64, 64, lane, 0);
      v16h a1A = ldsA(h1h,      64, lane, 1);
      v16h a1B = ldsA(h1h + 16*64, 64, lane, 1);
      accA = WMMA_F16(axA, Bx0,   accA);
      accB = WMMA_F16(axB, Bx0,   accB);
      accA = WMMA_F16(a0A, Bh0_0, accA);
      accB = WMMA_F16(a0B, Bh0_0, accB);
      accA = WMMA_F16(a1A, Bh0_1, accA);
      accB = WMMA_F16(a1B, Bh0_1, accB);
#pragma unroll
      for (int r = 0; r < 8; r++) {
        zbuf[(mbase + r) * 160 + col0 + ncol]      = accA[r];
        zbuf[(mbase + r + 16) * 160 + col0 + ncol] = accB[r];
      }
    }
    __syncthreads();

    // ---- layer0 elementwise: gates -> c1,h1 (straight-line, 4 units/thread) ----
#pragma unroll
    for (int k = 0; k < 4; k++) cell_update(zbuf, c1, h1f, h1h, er[k], ej[k]);
    __syncthreads();

    // ---- layer1 GEMM: z1 = h1 @ Wih1^T + h2 @ Whh1^T + bias ----
    {
      v8f accA, accB;
#pragma unroll
      for (int r = 0; r < 8; r++) { accA[r] = bias1; accB[r] = bias1; }
      v16h a0A = ldsA(h1h,      64, lane, 0);
      v16h a0B = ldsA(h1h + 16*64, 64, lane, 0);
      v16h a1A = ldsA(h1h,      64, lane, 1);
      v16h a1B = ldsA(h1h + 16*64, 64, lane, 1);
      v16h a2A = ldsA(h2h,      64, lane, 0);
      v16h a2B = ldsA(h2h + 16*64, 64, lane, 0);
      v16h a3A = ldsA(h2h,      64, lane, 1);
      v16h a3B = ldsA(h2h + 16*64, 64, lane, 1);
      accA = WMMA_F16(a0A, Bi1_0, accA);
      accB = WMMA_F16(a0B, Bi1_0, accB);
      accA = WMMA_F16(a1A, Bi1_1, accA);
      accB = WMMA_F16(a1B, Bi1_1, accB);
      accA = WMMA_F16(a2A, Bh1_0, accA);
      accB = WMMA_F16(a2B, Bh1_0, accB);
      accA = WMMA_F16(a3A, Bh1_1, accA);
      accB = WMMA_F16(a3B, Bh1_1, accB);
#pragma unroll
      for (int r = 0; r < 8; r++) {
        zbuf[(mbase + r) * 160 + col0 + ncol]      = accA[r];
        zbuf[(mbase + r + 16) * 160 + col0 + ncol] = accB[r];
      }
    }
    __syncthreads();

    // ---- layer1 elementwise: gates -> c2,h2 ----
#pragma unroll
    for (int k = 0; k < 4; k++) cell_update(zbuf, c2, h2f, h2h, er[k], ej[k]);
    __syncthreads();

    // ---- heads: hp = interleave(h1,h2); softplus(hp @ W^T + b), rolled time ----
    int tout = t + PRED; if (tout >= WSTEP) tout -= WSTEP;   // (t - SEQ) mod W
#pragma unroll
    for (int un = 0; un < 3; un++) {
      if (un == 2 && tid >= TILE_M * 21 - 2 * NTHR) break;   // third unit: tid<32 only
      int row = hr[un];
      int s   = hs[un];
      float acc = bhead[s];
      const float4* h1v = (const float4*)(h1f + row * HID);  // 160B-aligned
      const float4* h2v = (const float4*)(h2f + row * HID);
      const float4* wv  = (const float4*)(whead + s * 80);   // 320B-aligned
#pragma unroll
      for (int q = 0; q < 10; q++) {                          // ds_load_b128 stream
        float4 a  = h1v[q];
        float4 b  = h2v[q];
        float4 w0 = wv[2 * q];
        float4 w1 = wv[2 * q + 1];
        // hp[2j]=h1[j], hp[2j+1]=h2[j]
        acc += a.x * w0.x + b.x * w0.y + a.y * w0.z + b.y * w0.w
             + a.z * w1.x + b.z * w1.y + a.w * w1.z + b.w * w1.w;
      }
      float v = softplusf(acc);
      int b = b0 + row;
      if (s == 0) out[(size_t)tout * B_TOT + b] = v;                                        // beta_0
      else        out[(size_t)WSTEP * B_TOT + ((size_t)tout * B_TOT + b) * NS + (s - 1)] = v; // gamma
    }
    // next iter's first __syncthreads (after x load) separates head reads from h-updates
  }
}

extern "C" void kernel_launch(void* const* d_in, const int* in_sizes, int n_in,
                              void* d_out, int out_size, void* d_ws, size_t ws_size,
                              hipStream_t stream) {
  const float* x_enc = (const float*)d_in[0];
  // d_in[1]=x_mark_enc, d_in[2]=x_dec, d_in[4]=x_mark_dec : unused by reference math
  const float* y_enc = (const float*)d_in[3];
  const float* w_ih0 = (const float*)d_in[5];
  const float* w_hh0 = (const float*)d_in[6];
  const float* b_ih0 = (const float*)d_in[7];
  const float* b_hh0 = (const float*)d_in[8];
  const float* w_ih1 = (const float*)d_in[9];
  const float* w_hh1 = (const float*)d_in[10];
  const float* b_ih1 = (const float*)d_in[11];
  const float* b_hh1 = (const float*)d_in[12];
  const float* w_b0  = (const float*)d_in[13];
  const float* b_b0  = (const float*)d_in[14];
  const float* w_g   = (const float*)d_in[15];
  const float* b_g   = (const float*)d_in[16];
  float* out = (float*)d_out;

  dim3 grid(B_TOT / TILE_M);   // 128 persistent batch tiles
  dim3 block(NTHR);            // 10 wave32s
  lstm_deepar_kernel<<<grid, block, 0, stream>>>(
      x_enc, y_enc, w_ih0, w_hh0, b_ih0, b_hh0,
      w_ih1, w_hh1, b_ih1, b_hh1, w_b0, b_b0, w_g, b_g, out);
}
